// Block_47193100649015
// MI455X (gfx1250) — matrix operand order repaired and checked by reading, not compiled, not run on hardware
//
#include <hip/hip_runtime.h>
#include <hip/hip_bf16.h>
#include <math.h>

// ---------------------------------------------------------------------------
// Types for CDNA5 WMMA (wave32): v_wmma_f32_16x16x32_bf16
// ---------------------------------------------------------------------------
typedef __attribute__((ext_vector_type(16))) __bf16         v16bf;
typedef __attribute__((ext_vector_type(8)))  float          v8f;
typedef __attribute__((ext_vector_type(8)))  unsigned short ushort8;

union FragBF {
  v16bf          v;
  ushort8        h[2];
  unsigned short s[16];
};

__device__ __forceinline__ unsigned short f32_to_bf16(float f) {
  unsigned u = __float_as_uint(f);
  u += 0x7FFFu + ((u >> 16) & 1u);   // round-to-nearest-even
  return (unsigned short)(u >> 16);
}

__device__ __forceinline__ v8f wmma_bf16(const FragBF& a, const FragBF& b, v8f c) {
  return __builtin_amdgcn_wmma_f32_16x16x32_bf16(
      false, a.v, false, b.v, (short)0, c, false, false);
}

// ---------------------------------------------------------------------------
// Tensor Data Mover staging (gfx1250).  2D tile load, bf16 elements, LDS rows
// padded 32 halves -> 40 halves (pad_interval=16 DWORDs, pad_amount=4 DWORDs).
// ---------------------------------------------------------------------------
#if defined(__HIP_DEVICE_COMPILE__) && __has_builtin(__builtin_amdgcn_tensor_load_to_lds)
#define HAVE_TDM 1
typedef __attribute__((ext_vector_type(4))) unsigned int uint4v;
typedef __attribute__((ext_vector_type(8))) int          int8v;
typedef __attribute__((ext_vector_type(4))) int          int4v;

__device__ __forceinline__ void tdm_load_tile_2d(unsigned lds_off, const void* gaddr,
                                                 int tileK, int tileRows,
                                                 int K /*dim0 & row stride*/, int rowsTot) {
  unsigned long long ga = (unsigned long long)gaddr;
  uint4v g0;
  g0[0] = 1u;                                   // count=1 (valid user descriptor)
  g0[1] = lds_off;                              // lds_addr (bytes)
  g0[2] = (unsigned)(ga & 0xFFFFFFFFu);         // global_addr lo
  g0[3] = (unsigned)((ga >> 32) & 0x1FFFFFFu) | (2u << 30);  // global hi | type=2
  int8v g1;
  const unsigned pad = (1u << 20) | (3u << 22) | (3u << 25); // pad_en, 16 DW int., 4 DW amt
  g1[0] = (int)((1u << 16) | pad);              // data_size=1 (2B), wg_mask=0
  g1[1] = (int)((unsigned)(K & 0xFFFF) << 16);  // tensor_dim0[15:0]
  g1[2] = (int)(((unsigned)K >> 16) | ((unsigned)(rowsTot & 0xFFFF) << 16));
  g1[3] = (int)(((unsigned)rowsTot >> 16) | ((unsigned)tileK << 16));   // tile_dim0
  g1[4] = (int)((unsigned)tileRows);            // tile_dim1, tile_dim2=0
  g1[5] = (int)K;                               // tensor_dim0_stride lo
  g1[6] = 0;
  g1[7] = 0;
  const int4v z4 = {0, 0, 0, 0};
#if defined(__clang_major__) && (__clang_major__ >= 23)
  const int8v z8 = {0, 0, 0, 0, 0, 0, 0, 0};
  __builtin_amdgcn_tensor_load_to_lds(g0, g1, z4, z4, z8, 0);
#else
  __builtin_amdgcn_tensor_load_to_lds(g0, g1, z4, z4, 0);
#endif
}
#endif

// ---------------------------------------------------------------------------
// fp32 -> bf16 elementwise conversion
// ---------------------------------------------------------------------------
__global__ __launch_bounds__(256) void cvt_kernel(const float* __restrict__ in,
                                                  unsigned short* __restrict__ out,
                                                  long n) {
  long i = (long)blockIdx.x * 256 + threadIdx.x;
  if (i < n) out[i] = f32_to_bf16(in[i]);
}

// fp32 [K][N] -> bf16 transposed [N][K]  (one-time weight pre-transpose)
__global__ __launch_bounds__(256) void cvtT_kernel(const float* __restrict__ in,
                                                   unsigned short* __restrict__ out,
                                                   int K, int N) {
  long idx = (long)blockIdx.x * 256 + threadIdx.x;
  if (idx < (long)K * N) {
    int n = (int)(idx % N);
    int k = (int)(idx / N);
    out[(long)n * K + k] = f32_to_bf16(in[idx]);
  }
}

// ---------------------------------------------------------------------------
// LayerNorm: one block per row of N elements; optional f32 and bf16 outputs
// ---------------------------------------------------------------------------
__global__ __launch_bounds__(256) void layernorm_kernel(
    const float* __restrict__ x, const float* __restrict__ g,
    const float* __restrict__ b, float* __restrict__ outf,
    unsigned short* __restrict__ outb, int N) {
  __shared__ float red[256];
  __shared__ float s_mu, s_rstd;
  const int tid = threadIdx.x;
  const long row = blockIdx.x;
  const float* xr = x + row * (long)N;

  float s = 0.f;
  for (int i = tid; i < N; i += 256) s += xr[i];
  red[tid] = s; __syncthreads();
  for (int w = 128; w > 0; w >>= 1) { if (tid < w) red[tid] += red[tid + w]; __syncthreads(); }
  if (tid == 0) s_mu = red[0] / (float)N;
  __syncthreads();
  const float mu = s_mu;

  float v = 0.f;
  for (int i = tid; i < N; i += 256) { float d = xr[i] - mu; v += d * d; }
  red[tid] = v; __syncthreads();
  for (int w = 128; w > 0; w >>= 1) { if (tid < w) red[tid] += red[tid + w]; __syncthreads(); }
  if (tid == 0) s_rstd = rsqrtf(red[0] / (float)N + 1e-5f);
  __syncthreads();
  const float rstd = s_rstd;

  for (int i = tid; i < N; i += 256) {
    float y = (xr[i] - mu) * rstd * g[i] + b[i];
    if (outf) outf[row * (long)N + i] = y;
    if (outb) outb[row * (long)N + i] = f32_to_bf16(y);
  }
}

// ---------------------------------------------------------------------------
// Tiled bf16 WMMA GEMM: C[M,N] = A[M,K] @ W[K,N] + bias (+resid) (+gelu)
// A row-major [M][K]; WT is W PRE-TRANSPOSED, row-major [N][K].
// Block 256 threads = 8 waves; tile 128x128x32; each wave computes 64x32.
// Optional outputs: f32 C, bf16 C, and per-batch-transposed bf16 CbT[b][N][tT].
// ---------------------------------------------------------------------------
#define GBM 128
#define GBN 128
#define GBK 32
#define LDSP 40   // padded stride in halves (32 + 8): 80B, 16B aligned

__global__ __launch_bounds__(256) void gemm_bf16_kernel(
    const unsigned short* __restrict__ A, const unsigned short* __restrict__ WT,
    const float* __restrict__ bias, const float* __restrict__ resid,
    float* __restrict__ Cf, unsigned short* __restrict__ Cb,
    unsigned short* __restrict__ CbT, int tT,
    int M, int N, int K, int act) {
  __shared__ alignas(16) unsigned short As[GBM * LDSP];   // [row][k]
  __shared__ alignas(16) unsigned short BsT[GBN * LDSP];  // [n][k]

  const int tid  = threadIdx.x;
  const int wave = tid >> 5, lane = tid & 31;
  const int half = lane >> 4, l15 = lane & 15;
  const int waveM = wave >> 2;   // 0..1  (rows of 64)
  const int waveN = wave & 3;    // 0..3  (cols of 32)
  const long rowBase = (long)blockIdx.y * GBM;
  const long colBase = (long)blockIdx.x * GBN;

  v8f acc[4][2];
  const v8f vzero = {0.f, 0.f, 0.f, 0.f, 0.f, 0.f, 0.f, 0.f};
#pragma unroll
  for (int mi = 0; mi < 4; ++mi)
#pragma unroll
    for (int ni = 0; ni < 2; ++ni) acc[mi][ni] = vzero;

  const int arow = tid >> 1;          // 0..127
  const int acol = (tid & 1) * 16;    // 0 / 16

  for (int k0 = 0; k0 < K; k0 += GBK) {
#ifdef HAVE_TDM
    if (wave == 0) {
      tdm_load_tile_2d((unsigned)(unsigned long long)(void*)As,
                       A + rowBase * K + k0, GBK, GBM, K, M);
      tdm_load_tile_2d((unsigned)(unsigned long long)(void*)BsT,
                       WT + colBase * K + k0, GBK, GBN, K, N);
      __builtin_amdgcn_s_wait_tensorcnt(0);
    }
    __syncthreads();
#else
    {  // cooperative staging: both tiles are 128 rows x 32 contiguous halves
      const unsigned short* ga = A + (rowBase + arow) * (long)K + k0 + acol;
      ushort8 a0 = ((const ushort8*)ga)[0];
      ushort8 a1 = ((const ushort8*)ga)[1];
      const unsigned short* gb = WT + (colBase + arow) * (long)K + k0 + acol;
      ushort8 b0 = ((const ushort8*)gb)[0];
      ushort8 b1 = ((const ushort8*)gb)[1];
      if (k0 + GBK < K) {
        __builtin_prefetch(A + (rowBase + arow) * (long)K + k0 + GBK + acol, 0, 0);
        __builtin_prefetch(WT + (colBase + arow) * (long)K + k0 + GBK + acol, 0, 0);
      }
      ((ushort8*)(As + arow * LDSP + acol))[0] = a0;
      ((ushort8*)(As + arow * LDSP + acol))[1] = a1;
      ((ushort8*)(BsT + arow * LDSP + acol))[0] = b0;
      ((ushort8*)(BsT + arow * LDSP + acol))[1] = b1;
    }
    __syncthreads();
#endif

    // ---- fragments + 4x2 WMMA ----
    FragBF afr[4], bfr[2];
#pragma unroll
    for (int mi = 0; mi < 4; ++mi) {
      const int m = waveM * 64 + mi * 16 + l15;
      afr[mi].h[0] = *(const ushort8*)(As + m * LDSP + half * 8);        // K 0..7 / 8..15
      afr[mi].h[1] = *(const ushort8*)(As + m * LDSP + 16 + half * 8);   // K 16..23 / 24..31
    }
#pragma unroll
    for (int ni = 0; ni < 2; ++ni) {
      const int n = waveN * 32 + ni * 16 + l15;
      bfr[ni].h[0] = *(const ushort8*)(BsT + n * LDSP + half * 8);       // K 0..7 / 8..15
      bfr[ni].h[1] = *(const ushort8*)(BsT + n * LDSP + 16 + half * 8);  // K 16..23 / 24..31
    }
#pragma unroll
    for (int mi = 0; mi < 4; ++mi)
#pragma unroll
      for (int ni = 0; ni < 2; ++ni)
        acc[mi][ni] = wmma_bf16(afr[mi], bfr[ni], acc[mi][ni]);
    __syncthreads();
  }

  // ---- epilogue: bias (+gelu) (+residual) -> f32 / bf16 / transposed bf16 ----
#pragma unroll
  for (int mi = 0; mi < 4; ++mi)
#pragma unroll
    for (int ni = 0; ni < 2; ++ni) {
      const long row0 = rowBase + waveM * 64 + mi * 16 + half * 8;
      const long col  = colBase + waveN * 32 + ni * 16 + l15;
      const float bia = bias ? bias[col] : 0.f;
#pragma unroll
      for (int r = 0; r < 8; ++r) {
        const long row = row0 + r;
        float val = acc[mi][ni][r] + bia;
        if (act) val = 0.5f * val * (1.0f + erff(val * 0.70710678118654752f));
        if (resid) val += resid[row * N + col];
        if (Cf) Cf[row * N + col] = val;
        if (Cb) Cb[row * N + col] = f32_to_bf16(val);
        if (CbT) {
          const long bidx = row / tT, trow = row % tT;
          CbT[(bidx * N + col) * tT + trow] = f32_to_bf16(val);
        }
      }
    }
}

// ---------------------------------------------------------------------------
// Flash attention (bf16 in/out, f32 online softmax).
// Q/K/Out: [B][T][768] row-major, head slice at h*64.
// V: PRE-TRANSPOSED per batch: [B][768][Tk]  (written by the V-proj GEMM).
// Block: 128 threads = 4 waves; each wave owns 16 query rows of a 64-row tile.
// ---------------------------------------------------------------------------
__global__ __launch_bounds__(128) void flash_attn_kernel(
    const unsigned short* __restrict__ Q, const unsigned short* __restrict__ Kt,
    const unsigned short* __restrict__ vT, unsigned short* __restrict__ Out,
    int Tq, int Tk, int causal) {
  const int LD = 768;
  __shared__ alignas(16) unsigned short Pl[4][16][LDSP];

  const int tid = threadIdx.x, wave = tid >> 5, lane = tid & 31;
  const int half = lane >> 4, l15 = lane & 15;
  const int h = blockIdx.y, b = blockIdx.z;
  const long baseQ  = (long)b * Tq * LD + h * 64;
  const long baseKV = (long)b * Tk * LD + h * 64;
  const int qblk = blockIdx.x * 64;
  const int qrow0 = qblk + wave * 16;

  FragBF qf[2];
  {
    const int m = qrow0 + l15;
#pragma unroll
    for (int c = 0; c < 2; ++c) {
      const unsigned short* p = Q + baseQ + (long)m * LD + c * 32;
      qf[c].h[0] = *(const ushort8*)(p + half * 8);
      qf[c].h[1] = *(const ushort8*)(p + 16 + half * 8);
    }
  }

  const v8f vzero = {0.f, 0.f, 0.f, 0.f, 0.f, 0.f, 0.f, 0.f};
  v8f o_[4];
#pragma unroll
  for (int a = 0; a < 4; ++a) o_[a] = vzero;
  float mx[8], li[8];
#pragma unroll
  for (int r = 0; r < 8; ++r) { mx[r] = -1e30f; li[r] = 0.f; }

  const int kend = causal ? (qblk + 64) : Tk;
  for (int kb = 0; kb < kend; kb += 32) {
    // ---- S = Q @ K^T : 16 queries x 32 keys ----
    v8f s_[2] = {vzero, vzero};
#pragma unroll
    for (int c = 0; c < 2; ++c) {
#pragma unroll
      for (int nt = 0; nt < 2; ++nt) {
        FragBF kf;
        const int key = kb + nt * 16 + l15;
        const unsigned short* kp = Kt + baseKV + (long)key * LD + c * 32 + half * 8;
        kf.h[0] = *(const ushort8*)kp;
        kf.h[1] = *(const ushort8*)(kp + 16);
        s_[nt] = wmma_bf16(qf[c], kf, s_[nt]);
      }
    }

    // ---- online softmax ----
#pragma unroll
    for (int r = 0; r < 8; ++r) {
      const int qg = qrow0 + half * 8 + r;
      float best = mx[r];
#pragma unroll
      for (int nt = 0; nt < 2; ++nt) {
        const int key = kb + nt * 16 + l15;
        float val = s_[nt][r] * 0.125f;
        if (causal && key > qg) val = -1e30f;
        s_[nt][r] = val;
        best = fmaxf(best, val);
      }
#pragma unroll
      for (int off = 1; off < 16; off <<= 1)
        best = fmaxf(best, __shfl_xor(best, off, 32));
      const float mnew = best;
      const float alpha = __expf(mx[r] - mnew);
      float rs = 0.f;
#pragma unroll
      for (int nt = 0; nt < 2; ++nt) {
        float p = __expf(s_[nt][r] - mnew);
        s_[nt][r] = p;
        rs += p;
      }
#pragma unroll
      for (int off = 1; off < 16; off <<= 1) rs += __shfl_xor(rs, off, 32);
      li[r] = li[r] * alpha + rs;
      mx[r] = mnew;
#pragma unroll
      for (int a = 0; a < 4; ++a) o_[a][r] *= alpha;
    }

    // ---- P: C-layout -> A-fragment via LDS ----
#pragma unroll
    for (int nt = 0; nt < 2; ++nt)
#pragma unroll
      for (int r = 0; r < 8; ++r)
        Pl[wave][half * 8 + r][nt * 16 + l15] = f32_to_bf16(s_[nt][r]);
    __syncthreads();

    FragBF pf;
    pf.h[0] = *(const ushort8*)(&Pl[wave][l15][half * 8]);
    pf.h[1] = *(const ushort8*)(&Pl[wave][l15][16 + half * 8]);

    // ---- O += P @ V  (V transposed: contiguous over keys) ----
#pragma unroll
    for (int nt2 = 0; nt2 < 4; ++nt2) {
      FragBF vf;
      const unsigned short* vp =
          vT + ((long)b * LD + h * 64 + nt2 * 16 + l15) * Tk + kb + half * 8;
      vf.h[0] = *(const ushort8*)vp;
      vf.h[1] = *(const ushort8*)(vp + 16);
      o_[nt2] = wmma_bf16(pf, vf, o_[nt2]);
    }
    __syncthreads();
  }

#pragma unroll
  for (int nt2 = 0; nt2 < 4; ++nt2)
#pragma unroll
    for (int r = 0; r < 8; ++r) {
      const long row = qrow0 + half * 8 + r;
      const float inv = 1.0f / li[r];
      Out[baseQ + row * LD + nt2 * 16 + l15] = f32_to_bf16(o_[nt2][r] * inv);
    }
}

// ---------------------------------------------------------------------------
// Host orchestration
// ---------------------------------------------------------------------------
extern "C" void kernel_launch(void* const* d_in, const int* in_sizes, int n_in,
                              void* d_out, int out_size, void* d_ws, size_t ws_size,
                              hipStream_t stream) {
  const int NINP = 768, PROT = 1024, HID = 3072, HEADS = 12;
  const int Bb = 4, T = 2048, S = 512;
  const long MT = (long)Bb * T;
  const long MS = (long)Bb * S;

  const float* x    = (const float*)d_in[0];
  const float* enc  = (const float*)d_in[1];
  const float* ln1g = (const float*)d_in[2];
  const float* ln1b = (const float*)d_in[3];
  const float* ln2g = (const float*)d_in[4];
  const float* ln2b = (const float*)d_in[5];
  const float* sWq = (const float*)d_in[6],  *sbq = (const float*)d_in[7];
  const float* sWk = (const float*)d_in[8],  *sbk = (const float*)d_in[9];
  const float* sWv = (const float*)d_in[10], *sbv = (const float*)d_in[11];
  const float* sWo = (const float*)d_in[12], *sbo = (const float*)d_in[13];
  const float* cWq = (const float*)d_in[14], *cbq = (const float*)d_in[15];
  const float* cWk = (const float*)d_in[16], *cbk = (const float*)d_in[17];
  const float* cWv = (const float*)d_in[18], *cbv = (const float*)d_in[19];
  const float* cWo = (const float*)d_in[20], *cbo = (const float*)d_in[21];
  const float* mW1 = (const float*)d_in[22], *mb1 = (const float*)d_in[23];
  const float* mW2 = (const float*)d_in[24], *mb2 = (const float*)d_in[25];
  float* out = (float*)d_out;

  char* ws = (char*)d_ws;
  size_t off = 0;
  auto carve = [&](size_t bytes) -> char* {
    char* p = ws + off;
    off += (bytes + 255) & ~(size_t)255;
    return p;
  };
  typedef unsigned short us;
  // weights, pre-transposed to [N][K] bf16
  us* wsWq = (us*)carve((size_t)NINP * NINP * 2);
  us* wsWk = (us*)carve((size_t)NINP * NINP * 2);
  us* wsWv = (us*)carve((size_t)NINP * NINP * 2);
  us* wsWo = (us*)carve((size_t)NINP * NINP * 2);
  us* wcWq = (us*)carve((size_t)NINP * NINP * 2);
  us* wcWk = (us*)carve((size_t)PROT * NINP * 2);
  us* wcWv = (us*)carve((size_t)PROT * NINP * 2);
  us* wcWo = (us*)carve((size_t)NINP * NINP * 2);
  us* wmW1 = (us*)carve((size_t)NINP * HID * 2);
  us* wmW2 = (us*)carve((size_t)HID * NINP * 2);
  float* xn_f  = (float*)carve((size_t)MT * NINP * 4);
  us*    xn_b  = (us*)carve((size_t)MT * NINP * 2);
  us*    qb    = (us*)carve((size_t)MT * NINP * 2);
  us*    kb    = (us*)carve((size_t)MT * NINP * 2);
  us*    vTb   = (us*)carve((size_t)MT * NINP * 2);  // V transposed [B][768][Tk]
  us*    yb    = (us*)carve((size_t)MT * NINP * 2);
  float* x1_f  = (float*)carve((size_t)MT * NINP * 4);
  us*    x1_b  = (us*)carve((size_t)MT * NINP * 2);
  us*    enc_b = (us*)carve((size_t)MS * PROT * 2);
  float* x2_f  = (float*)carve((size_t)MT * NINP * 4);
  us*    h_b   = (us*)carve((size_t)MT * NINP * 2);
  us*    h1_b  = (us*)carve((size_t)MT * HID * 2);
  (void)ws_size; (void)in_sizes; (void)n_in; (void)out_size;

  auto cvt = [&](const float* src, us* dst, long n) {
    cvt_kernel<<<dim3((unsigned)((n + 255) / 256)), dim3(256), 0, stream>>>(src, dst, n);
  };
  auto cvtT = [&](const float* src, us* dst, int K, int N) {
    long n = (long)K * N;
    cvtT_kernel<<<dim3((unsigned)((n + 255) / 256)), dim3(256), 0, stream>>>(src, dst, K, N);
  };
  auto gemm = [&](const us* A, const us* WT, const float* bias, const float* resid,
                  float* Cf, us* Cb, us* CbT, int tT, int M, int N, int K, int act) {
    dim3 grid((unsigned)(N / GBN), (unsigned)(M / GBM));
    gemm_bf16_kernel<<<grid, dim3(256), 0, stream>>>(A, WT, bias, resid, Cf, Cb, CbT, tT,
                                                     M, N, K, act);
  };

  // weights -> bf16 transposed [N][K]; activations -> bf16
  cvtT(sWq, wsWq, NINP, NINP);  cvtT(sWk, wsWk, NINP, NINP);
  cvtT(sWv, wsWv, NINP, NINP);  cvtT(sWo, wsWo, NINP, NINP);
  cvtT(cWq, wcWq, NINP, NINP);  cvtT(cWk, wcWk, PROT, NINP);
  cvtT(cWv, wcWv, PROT, NINP);  cvtT(cWo, wcWo, NINP, NINP);
  cvtT(mW1, wmW1, NINP, HID);   cvtT(mW2, wmW2, HID, NINP);
  cvt(enc, enc_b, (long)MS * PROT);

  // xn = ln1(x)
  layernorm_kernel<<<dim3((unsigned)MT), dim3(256), 0, stream>>>(x, ln1g, ln1b, xn_f, xn_b, NINP);

  // self-attn projections (V written transposed for the P@V WMMA path)
  gemm(xn_b, wsWq, sbq, nullptr, nullptr, qb, nullptr, 0, (int)MT, NINP, NINP, 0);
  gemm(xn_b, wsWk, sbk, nullptr, nullptr, kb, nullptr, 0, (int)MT, NINP, NINP, 0);
  gemm(xn_b, wsWv, sbv, nullptr, nullptr, nullptr, vTb, T, (int)MT, NINP, NINP, 0);
  flash_attn_kernel<<<dim3(T / 64, HEADS, Bb), dim3(128), 0, stream>>>(qb, kb, vTb, yb, T, T, 1);
  // x1 = xn + y @ Wo + bo
  gemm(yb, wsWo, sbo, xn_f, x1_f, x1_b, nullptr, 0, (int)MT, NINP, NINP, 0);

  // cross-attn projections
  gemm(x1_b, wcWq, cbq, nullptr, nullptr, qb, nullptr, 0, (int)MT, NINP, NINP, 0);
  gemm(enc_b, wcWk, cbk, nullptr, nullptr, kb, nullptr, 0, (int)MS, NINP, PROT, 0);
  gemm(enc_b, wcWv, cbv, nullptr, nullptr, nullptr, vTb, S, (int)MS, NINP, PROT, 0);
  flash_attn_kernel<<<dim3(T / 64, HEADS, Bb), dim3(128), 0, stream>>>(qb, kb, vTb, yb, T, S, 0);
  // x2 = x1 + y2 @ cWo + cbo
  gemm(yb, wcWo, cbo, x1_f, x2_f, nullptr, nullptr, 0, (int)MT, NINP, NINP, 0);

  // MLP with fused GELU, final residual into d_out
  layernorm_kernel<<<dim3((unsigned)MT), dim3(256), 0, stream>>>(x2_f, ln2g, ln2b, nullptr, h_b, NINP);
  gemm(h_b, wmW1, mb1, nullptr, nullptr, h1_b, nullptr, 0, (int)MT, HID, NINP, 1);
  gemm(h1_b, wmW2, mb2, x2_f, out, nullptr, nullptr, 0, (int)MT, NINP, HID, 0);
}
